// CellMatesTransformer_24120536334929
// MI455X (gfx1250) — matrix operand order repaired and checked by reading, not compiled, not run on hardware
//
#include <hip/hip_runtime.h>
#include <hip/hip_bf16.h>
#include <math.h>

typedef __attribute__((ext_vector_type(16))) __bf16 v16bf;
typedef __attribute__((ext_vector_type(8)))  __bf16 bf16x8;
typedef __attribute__((ext_vector_type(2)))  __bf16 bf16x2;
typedef __attribute__((ext_vector_type(8)))  float  v8f;
typedef __attribute__((ext_vector_type(4)))  float  f32x4;
typedef int v4i __attribute__((vector_size(4 * sizeof(int))));

#define EPI_NONE 0
#define EPI_GELU 1

#define AS1 __attribute__((address_space(1)))
#define AS3 __attribute__((address_space(3)))

#if defined(__gfx1250__) && __has_builtin(__builtin_amdgcn_global_load_async_to_lds_b128) && __has_builtin(__builtin_amdgcn_s_wait_asynccnt)
#define USE_ASYNC_LDS 1
#endif

// ---------------------------------------------------------------------------
// Batched WMMA GEMM:  C[m,n] = epi( sum_k A[m,k]*B[k,n] + bias[n] )
//   A[bb*Ab + hh*Ah + m*ldA + k]   fp32, k-contiguous (all call sites)
//   B[bb*Bb + hh*Bh + k*sBK + n*sBN] fp32, arbitrary strides
//   C[bb*Cb + hh*Ch + m*ldC + n]   fp32
// 256 threads = 8 waves; block tile 128(M) x 64(N); K step 32.
// Wave w owns rows [16w,16w+16) x all 64 N columns: 1 A fragment feeds
// 4 v_wmma_f32_16x16x32_bf16 per K step. A tile staged raw f32 in LDS via
// async global->LDS b128 copies (ASYNCcnt); B tile converted to bf16 on
// store so fragments are pure ds_load_b128.
// ---------------------------------------------------------------------------
template<int EPI>
__global__ __launch_bounds__(256)
void gemm_bf16_wmma(const float* __restrict__ A, long ldA, long Ab, long Ah,
                    const float* __restrict__ Bm, long sBK, long sBN, long Bb, long Bh,
                    const float* __restrict__ bias,
                    float* __restrict__ C, long ldC, long Cb, long Ch,
                    int M, int N, int Kd, int Hbat)
{
  __shared__ float  Af[128][36];   // [m][k] f32, rows 144B (16B aligned)
  __shared__ __bf16 Bs[64][40];    // [n][k] bf16, rows 80B (16B aligned)

  const int g  = blockIdx.z;
  const int bb = g / Hbat, hh = g - bb * Hbat;
  A  += (long)bb * Ab + (long)hh * Ah;
  Bm += (long)bb * Bb + (long)hh * Bh;
  C  += (long)bb * Cb + (long)hh * Ch;
  const int m0   = blockIdx.y * 128;
  const int n0   = blockIdx.x * 64;
  const int tid  = threadIdx.x;
  const int wave = tid >> 5, lane = tid & 31;
  const int row  = lane & 15, hf = lane >> 4;
  const int wm   = wave * 16;

  v8f acc[4] = {};

  for (int k0 = 0; k0 < Kd; k0 += 32) {
    // ---- A tile: 128x32 f32 = 1024 x 16B transfers ----
#if USE_ASYNC_LDS
#pragma unroll
    for (int it = 0; it < 4; ++it) {
      const int p = tid + it * 256;
      const int r = p >> 3, gq = (p & 7) * 4;
      __builtin_amdgcn_global_load_async_to_lds_b128(
          (AS1 v4i*)(A + (long)(m0 + r) * ldA + (k0 + gq)),
          (AS3 v4i*)&Af[r][gq], 0, 0);
    }
#else
#pragma unroll
    for (int it = 0; it < 4; ++it) {
      const int p = tid + it * 256;
      const int r = p >> 3, gq = (p & 7) * 4;
      *(f32x4*)&Af[r][gq] = *(const f32x4*)(A + (long)(m0 + r) * ldA + (k0 + gq));
    }
#endif
    // ---- B tile: 32x64 -> bf16 LDS [n][k] (handles arbitrary strides) ----
#pragma unroll
    for (int it = 0; it < 4; ++it) {
      const int p = tid + it * 256;           // pair index 0..1023
      const int n = p >> 4, kk = 2 * (p & 15);
      const float x0 = Bm[(long)(k0 + kk)     * sBK + (long)(n0 + n) * sBN];
      const float x1 = Bm[(long)(k0 + kk + 1) * sBK + (long)(n0 + n) * sBN];
      bf16x2 pk; pk[0] = (__bf16)x0; pk[1] = (__bf16)x1;
      *(bf16x2*)&Bs[n][kk] = pk;
    }
#if USE_ASYNC_LDS
    __builtin_amdgcn_s_wait_asynccnt(0);
#endif
    __syncthreads();

    // ---- A fragment (ISA 16-bit A 16x32 layout): two 8-elem K runs ----
    v16bf af;
    {
      const f32x4 a0 = *(const f32x4*)&Af[wm + row][8 * hf];
      const f32x4 a1 = *(const f32x4*)&Af[wm + row][8 * hf + 4];
      const f32x4 a2 = *(const f32x4*)&Af[wm + row][16 + 8 * hf];
      const f32x4 a3 = *(const f32x4*)&Af[wm + row][20 + 8 * hf];
#pragma unroll
      for (int i = 0; i < 4; ++i) {
        af[i]      = (__bf16)a0[i];
        af[4 + i]  = (__bf16)a1[i];
        af[8 + i]  = (__bf16)a2[i];
        af[12 + i] = (__bf16)a3[i];
      }
    }
    // ---- 4 B fragments (two 8-elem K runs each) + 4 WMMA ----
#pragma unroll
    for (int c = 0; c < 4; ++c) {
      const bf16x8 b0 = *(const bf16x8*)&Bs[16 * c + row][8 * hf];
      const bf16x8 b1 = *(const bf16x8*)&Bs[16 * c + row][16 + 8 * hf];
      const v16bf bfr = __builtin_shufflevector(b0, b1, 0, 1, 2, 3, 4, 5, 6, 7,
                                                8, 9, 10, 11, 12, 13, 14, 15);
      acc[c] = __builtin_amdgcn_wmma_f32_16x16x32_bf16(false, af, false, bfr,
                                                       (short)0, acc[c], false, false);
    }
    __syncthreads();
  }

  // C/D layout: lanes0-15 -> M=i, lanes16-31 -> M=8+i ; N = 16c + (lane&15)
#pragma unroll
  for (int c = 0; c < 4; ++c) {
    const int n  = n0 + 16 * c + row;
    const float bv = bias ? bias[n] : 0.0f;
#pragma unroll
    for (int i = 0; i < 8; ++i) {
      const int m = m0 + wm + i + 8 * hf;
      float v = acc[c][i] + bv;
      if (EPI == EPI_GELU) v = 0.5f * v * (1.0f + erff(v * 0.7071067811865475f));
      C[(long)m * ldC + n] = v;
    }
  }
}

// ---------------------------------------------------------------------------
// Elementwise / gather / reduction kernels
// ---------------------------------------------------------------------------
__global__ void embed_kernel(const int* __restrict__ ct, const float* __restrict__ emb,
                             float* __restrict__ h, int total, int Dd) {
  int i = blockIdx.x * blockDim.x + threadIdx.x;
  if (i < total) h[i] = emb[ct[i / Dd] * Dd + (i % Dd)];
}

__global__ void didx_kernel(const float* __restrict__ dist, int* __restrict__ didx, int total) {
  int i = blockIdx.x * blockDim.x + threadIdx.x;
  if (i >= total) return;
  float v = dist[i];
  int idx = 0;
#pragma unroll
  for (int t = 1; t <= 14; ++t) idx += (10.0f * (float)t < v) ? 1 : 0;  // searchsorted left
  didx[i] = idx;
}

__global__ void dr_kernel(const int* __restrict__ didx, int* __restrict__ dr, int Bn, int Ln) {
  int i = blockIdx.x * blockDim.x + threadIdx.x;
  if (i >= Bn * Ln) return;
  int b = i / Ln, x = i - b * Ln;
  dr[i] = didx[(long)b * Ln * Ln + x];            // didx[:,0] row
}

// T[b,l,h, tab*15+bin] = sum_k Q[b,l,h,k] * dist_emb[tab][bin][k], tab in {0,1,2}
__global__ void tproj_kernel(const float* __restrict__ Q, const float* __restrict__ de,
                             float* __restrict__ T, int total, int Hn, int Kn) {
  int i = blockIdx.x * blockDim.x + threadIdx.x;
  if (i >= total) return;
  int j = i % 45; long blh = i / 45;
  const float* dev = de + (long)j * Kn;
  const float* q = Q + (blh / Hn) * (long)(Hn * Kn) + (blh % Hn) * Kn;
  float s = 0.0f;
  for (int k = 0; k < Kn; ++k) s += q[k] * dev[k];
  T[i] = s;
}

// Fused score assembly + mask + softmax, one block per (b,h,l) row of length L.
__global__ __launch_bounds__(384)
void score_softmax_kernel(const float* __restrict__ E, const float* __restrict__ T,
                          const int* __restrict__ didx, const int* __restrict__ dr,
                          const float* __restrict__ mask, float* __restrict__ Aout,
                          int Ln, int Hn) {
  const int g = blockIdx.x;                       // (b*H + h)*L + l
  const int l = g % Ln; const int bh = g / Ln;
  const int b = bh / Hn;
  const int x = threadIdx.x;
  const float* Trow = T + ((long)(b * Ln + l) * Hn + (bh % Hn)) * 45;
  const int drbl = dr[b * Ln + l];

  float e = E[(long)g * Ln + x];
  e += Trow[didx[((long)b * Ln + l) * Ln + x]];   // de0[didx]
  e += Trow[15 + drbl];                           // de1[dr[b,l]] (row bias)
  e += Trow[30 + dr[b * Ln + x]];                 // de2[dr[b,x]]
  const bool valid = (mask[b * Ln + x] != 0.0f);
  float ev = valid ? e : -__builtin_inff();

  __shared__ float red[12];
  __shared__ float stat[2];
  float m = ev;
  for (int o = 16; o; o >>= 1) m = fmaxf(m, __shfl_xor(m, o, 32));
  if ((x & 31) == 0) red[x >> 5] = m;
  __syncthreads();
  if (x == 0) { float t = red[0]; for (int i = 1; i < 12; ++i) t = fmaxf(t, red[i]); stat[0] = t; }
  __syncthreads();
  float p = valid ? __expf(ev - stat[0]) : 0.0f;
  float s = p;
  for (int o = 16; o; o >>= 1) s += __shfl_xor(s, o, 32);
  if ((x & 31) == 0) red[x >> 5] = s;
  __syncthreads();
  if (x == 0) { float t = 0.0f; for (int i = 0; i < 12; ++i) t += red[i]; stat[1] = t; }
  __syncthreads();
  Aout[(long)g * Ln + x] = p / stat[1];
}

// S[g, 0..14]  = sum_x A[g,x] * [didx[b,l,x]==j]
// S[g, 15..29] = sum_x A[g,x] * [dr[b,x]==j]
__global__ __launch_bounds__(384)
void saccum_kernel(const float* __restrict__ Ap, const int* __restrict__ didx,
                   const int* __restrict__ dr, float* __restrict__ S, int Ln, int Hn) {
  __shared__ float ls[30];
  const int g = blockIdx.x;
  const int l = g % Ln; const int bh = g / Ln; const int b = bh / Hn;
  const int x = threadIdx.x;
  if (x < 30) ls[x] = 0.0f;
  __syncthreads();
  float a = Ap[(long)g * Ln + x];
  atomicAdd(&ls[didx[((long)b * Ln + l) * Ln + x]], a);        // ds_add_f32
  atomicAdd(&ls[15 + dr[b * Ln + x]], a);
  __syncthreads();
  if (x < 30) S[(long)g * 30 + x] = ls[x];
}

// Z[b,l,h,k] (blhk for O-proj) = Zb[b,h,l,k] + S3@de3 + S5@de5 + de4[dr[b,l]]
__global__ void zextra_kernel(const float* __restrict__ Zb, const float* __restrict__ S,
                              const float* __restrict__ de3, const float* __restrict__ de4,
                              const float* __restrict__ de5, const int* __restrict__ dr,
                              float* __restrict__ Zout, int Ln, int Hn, int Kn, int total) {
  int i = blockIdx.x * blockDim.x + threadIdx.x;
  if (i >= total) return;
  const int k = i % Kn; int t = i / Kn;
  const int hh = t % Hn; const int bl = t / Hn;
  const int b = bl / Ln; const int l = bl - b * Ln;
  const long g = ((long)(b * Hn + hh) * Ln + l);
  const float* Srow = S + g * 30;
  float acc = Zb[g * Kn + k];
#pragma unroll
  for (int j = 0; j < 15; ++j)
    acc += Srow[j] * de3[j * Kn + k] + Srow[15 + j] * de5[j * Kn + k];
  acc += de4[dr[b * Ln + l] * Kn + k];
  Zout[((long)bl * Hn + hh) * Kn + k] = acc;
}

// h = LayerNorm(h + a) * g + b   (one block per token row, D=512)
__global__ __launch_bounds__(256)
void add_ln_kernel(const float* __restrict__ hin, const float* __restrict__ a,
                   const float* __restrict__ gm, const float* __restrict__ be,
                   float* __restrict__ hout, int Dd) {
  const long row = blockIdx.x;
  const int t = threadIdx.x;
  float v0 = hin[row * Dd + t]       + a[row * Dd + t];
  float v1 = hin[row * Dd + t + 256] + a[row * Dd + t + 256];
  __shared__ float red[8];
  __shared__ float stat[2];
  float s = v0 + v1;
  for (int o = 16; o; o >>= 1) s += __shfl_xor(s, o, 32);
  if ((t & 31) == 0) red[t >> 5] = s;
  __syncthreads();
  if (t == 0) { float q = 0.0f; for (int i = 0; i < 8; ++i) q += red[i]; stat[0] = q / (float)Dd; }
  __syncthreads();
  const float mu = stat[0];
  const float d0 = v0 - mu, d1 = v1 - mu;
  s = d0 * d0 + d1 * d1;
  for (int o = 16; o; o >>= 1) s += __shfl_xor(s, o, 32);
  if ((t & 31) == 0) red[t >> 5] = s;
  __syncthreads();
  if (t == 0) { float q = 0.0f; for (int i = 0; i < 8; ++i) q += red[i]; stat[1] = rsqrtf(q / (float)Dd + 1e-5f); }
  __syncthreads();
  const float inv = stat[1];
  hout[row * Dd + t]       = d0 * inv * gm[t]       + be[t];
  hout[row * Dd + t + 256] = d1 * inv * gm[t + 256] + be[t + 256];
}

__global__ void pool_kernel(const float* __restrict__ h, const float* __restrict__ mask,
                            float* __restrict__ pooled, int Bn, int Ln, int Dd) {
  int i = blockIdx.x * blockDim.x + threadIdx.x;
  if (i >= Bn * Dd) return;
  int b = i / Dd, d = i - b * Dd;
  float s = 0.0f;
  for (int l = 0; l < Ln; ++l) s += h[((long)b * Ln + l) * Dd + d] * mask[b * Ln + l];
  pooled[i] = s;
}

__global__ void mlp1_kernel(const float* __restrict__ pooled, const float* __restrict__ W,
                            const float* __restrict__ bi, float* __restrict__ hid,
                            int Dd, int Mn, int total) {
  int i = blockIdx.x * blockDim.x + threadIdx.x;
  if (i >= total) return;
  int b = i / Mn, m = i - b * Mn;
  float s = bi[m];
  for (int d = 0; d < Dd; ++d) s += pooled[b * Dd + d] * W[d * Mn + m];
  hid[i] = fmaxf(s, 0.0f);
}

__global__ __launch_bounds__(512)
void mlp2_kernel(const float* __restrict__ hid, const float* __restrict__ W,
                 const float* __restrict__ bi, float* __restrict__ out, int Mn) {
  const int b = blockIdx.x; const int t = threadIdx.x;
  float s = hid[b * Mn + t] * W[t];
  for (int o = 16; o; o >>= 1) s += __shfl_xor(s, o, 32);
  __shared__ float red[16];
  if ((t & 31) == 0) red[t >> 5] = s;
  __syncthreads();
  if (t == 0) { float q = 0.0f; for (int i = 0; i < 16; ++i) q += red[i]; out[b] = q + bi[0]; }
}

// ---------------------------------------------------------------------------
extern "C" void kernel_launch(void* const* d_in, const int* in_sizes, int n_in,
                              void* d_out, int out_size, void* d_ws, size_t ws_size,
                              hipStream_t stream) {
  enum { Bc = 4, Lc = 384, Dc = 512, Hc = 8, Kc = 64, Fc = 2048, Mc = 512, NDc = 15 };

  const int*   ct   = (const int*)  d_in[0];
  const float* dist = (const float*)d_in[1];
  const float* mask = (const float*)d_in[2];
  const float* cemb = (const float*)d_in[3];
  const float* demb = (const float*)d_in[4];
  const float* Wq = (const float*)d_in[5];  const float* bq = (const float*)d_in[6];
  const float* Wk = (const float*)d_in[7];  const float* bk = (const float*)d_in[8];
  const float* Wv = (const float*)d_in[9];  const float* bv = (const float*)d_in[10];
  const float* Wo = (const float*)d_in[11]; const float* bo = (const float*)d_in[12];
  const float* W1 = (const float*)d_in[13]; const float* b1 = (const float*)d_in[14];
  const float* W2 = (const float*)d_in[15]; const float* b2 = (const float*)d_in[16];
  const float* g1 = (const float*)d_in[17]; const float* be1 = (const float*)d_in[18];
  const float* g2 = (const float*)d_in[19]; const float* be2 = (const float*)d_in[20];
  const float* Wm1 = (const float*)d_in[21]; const float* bm1 = (const float*)d_in[22];
  const float* Wm2 = (const float*)d_in[23]; const float* bm2 = (const float*)d_in[24];
  float* out = (float*)d_out;
  (void)in_sizes; (void)n_in; (void)out_size; (void)ws_size;

  // workspace carve-out
  float* w = (float*)d_ws;
  float* h  = w; w += (size_t)Bc * Lc * Dc;
  float* Qb = w; w += (size_t)Bc * Lc * Dc;
  float* Kb = w; w += (size_t)Bc * Lc * Dc;
  float* Vb = w; w += (size_t)Bc * Lc * Dc;
  float* Eb = w; w += (size_t)Bc * Hc * Lc * Lc;   // scores, softmaxed in place
  float* Zb = w; w += (size_t)Bc * Hc * Lc * Kc;   // bhlk
  float* Zo = w; w += (size_t)Bc * Lc * Dc;        // blhk
  float* Tb = w; w += (size_t)Bc * Lc * Hc * 45;
  float* Sb = w; w += (size_t)Bc * Hc * Lc * 30;
  float* Ao = w; w += (size_t)Bc * Lc * Dc;
  float* F1 = w; w += (size_t)Bc * Lc * Fc;
  float* F2 = w; w += (size_t)Bc * Lc * Dc;
  float* Pool = w; w += (size_t)Bc * Dc;
  float* Hid  = w; w += (size_t)Bc * Mc;
  int* didx = (int*)w;
  int* drp  = didx + (size_t)Bc * Lc * Lc;

  const int MBL = Bc * Lc;                         // 1536 rows
  const long LD = (long)Lc * Dc;                   // per-b stride in token tensors

  embed_kernel<<<(Bc*Lc*Dc + 255)/256, 256, 0, stream>>>(ct, cemb, h, Bc*Lc*Dc, Dc);
  didx_kernel <<<(Bc*Lc*Lc + 255)/256, 256, 0, stream>>>(dist, didx, Bc*Lc*Lc);
  dr_kernel   <<<(Bc*Lc     + 255)/256, 256, 0, stream>>>(didx, drp, Bc, Lc);

  for (int l = 0; l < 2; ++l) {
    const float* Wq_l = Wq + (size_t)l*Dc*Dc;  const float* bq_l = bq + (size_t)l*Dc;
    const float* Wk_l = Wk + (size_t)l*Dc*Dc;  const float* bk_l = bk + (size_t)l*Dc;
    const float* Wv_l = Wv + (size_t)l*Dc*Dc;  const float* bv_l = bv + (size_t)l*Dc;
    const float* Wo_l = Wo + (size_t)l*Dc*Dc;  const float* bo_l = bo + (size_t)l*Dc;
    const float* W1_l = W1 + (size_t)l*Dc*Fc;  const float* b1_l = b1 + (size_t)l*Fc;
    const float* W2_l = W2 + (size_t)l*Fc*Dc;  const float* b2_l = b2 + (size_t)l*Dc;

    // Q/K/V projections: [1536x512] x [512x512], tiles 128x64
    dim3 gp(Dc/64, MBL/128, 1);
    gemm_bf16_wmma<EPI_NONE><<<gp, 256, 0, stream>>>(h, Dc, 0, 0, Wq_l, Dc, 1, 0, 0, bq_l, Qb, Dc, 0, 0, MBL, Dc, Dc, 1);
    gemm_bf16_wmma<EPI_NONE><<<gp, 256, 0, stream>>>(h, Dc, 0, 0, Wk_l, Dc, 1, 0, 0, bk_l, Kb, Dc, 0, 0, MBL, Dc, Dc, 1);
    gemm_bf16_wmma<EPI_NONE><<<gp, 256, 0, stream>>>(h, Dc, 0, 0, Wv_l, Dc, 1, 0, 0, bv_l, Vb, Dc, 0, 0, MBL, Dc, Dc, 1);

    // distance-bin projection tables (de0/de1/de2 contiguous: (3*15, K))
    tproj_kernel<<<(Bc*Lc*Hc*45 + 255)/256, 256, 0, stream>>>(Qb, demb, Tb, Bc*Lc*Hc*45, Hc, Kc);

    // E = Q K^T per (b,h): 32 batches of [384x64] x [64x384]
    dim3 ge(Lc/64, Lc/128, Bc*Hc);
    gemm_bf16_wmma<EPI_NONE><<<ge, 256, 0, stream>>>(
        Qb, Dc, LD, Kc,                 // A: Q[b*LD + h*64 + m*512 + k]
        Kb, 1, Dc, LD, Kc,              // B: Kh^T (k stride 1, x stride 512)
        nullptr,
        Eb, Lc, (long)Hc*Lc*Lc, (long)Lc*Lc,
        Lc, Lc, Kc, Hc);

    score_softmax_kernel<<<Bc*Hc*Lc, 384, 0, stream>>>(Eb, Tb, didx, drp, mask, Eb, Lc, Hc);
    saccum_kernel      <<<Bc*Hc*Lc, 384, 0, stream>>>(Eb, didx, drp, Sb, Lc, Hc);

    // Z = A V per (b,h): 32 batches of [384x384] x [384x64]
    dim3 gz(Kc/64, Lc/128, Bc*Hc);
    gemm_bf16_wmma<EPI_NONE><<<gz, 256, 0, stream>>>(
        Eb, Lc, (long)Hc*Lc*Lc, (long)Lc*Lc,
        Vb, Dc, 1, LD, Kc,
        nullptr,
        Zb, Kc, (long)Hc*Lc*Kc, (long)Lc*Kc,
        Lc, Kc, Lc, Hc);

    zextra_kernel<<<(Bc*Lc*Dc + 255)/256, 256, 0, stream>>>(
        Zb, Sb, demb + 3*NDc*Kc, demb + 4*NDc*Kc, demb + 5*NDc*Kc, drp, Zo,
        Lc, Hc, Kc, Bc*Lc*Dc);

    // O projection + residual LN
    gemm_bf16_wmma<EPI_NONE><<<gp, 256, 0, stream>>>(Zo, Dc, 0, 0, Wo_l, Dc, 1, 0, 0, bo_l, Ao, Dc, 0, 0, MBL, Dc, Dc, 1);
    add_ln_kernel<<<MBL, 256, 0, stream>>>(h, Ao, g1 + (size_t)l*Dc, be1 + (size_t)l*Dc, h, Dc);

    // FFN: gelu(h W1 + b1) W2 + b2, residual LN
    dim3 gf1(Fc/64, MBL/128, 1);
    gemm_bf16_wmma<EPI_GELU><<<gf1, 256, 0, stream>>>(h, Dc, 0, 0, W1_l, Fc, 1, 0, 0, b1_l, F1, Fc, 0, 0, MBL, Fc, Dc, 1);
    dim3 gf2(Dc/64, MBL/128, 1);
    gemm_bf16_wmma<EPI_NONE><<<gf2, 256, 0, stream>>>(F1, Fc, 0, 0, W2_l, Dc, 1, 0, 0, b2_l, F2, Dc, 0, 0, MBL, Dc, Fc, 1);
    add_ln_kernel<<<MBL, 256, 0, stream>>>(h, F2, g2 + (size_t)l*Dc, be2 + (size_t)l*Dc, h, Dc);
    (void)bk_l; (void)bv_l;
  }

  pool_kernel<<<(Bc*Dc + 255)/256, 256, 0, stream>>>(h, mask, Pool, Bc, Lc, Dc);
  mlp1_kernel<<<(Bc*Mc + 255)/256, 256, 0, stream>>>(Pool, Wm1, bm1, Hid, Dc, Mc, Bc*Mc);
  mlp2_kernel<<<Bc, 512, 0, stream>>>(Hid, Wm2, bm2, out, Mc);
}
